// FraudGAT_45071386804553
// MI455X (gfx1250) — compile-verified, matrix operand order checked by reading, not run on hardware
//
#include <hip/hip_runtime.h>
#include <stdint.h>

#define N_NODES 50000
#define N_EDGES 800000
#define E_TOT   (N_EDGES + N_NODES)
#define IN_CH   128
#define HID     64
#define HEADS   4
#define HW      256   // HEADS*HID
#define BN_EPS  1e-5f

typedef __attribute__((ext_vector_type(16))) __bf16 v16bf;
typedef __attribute__((ext_vector_type(8)))  float  v8f;

// ---------- helpers ----------
__device__ __forceinline__ unsigned short f2bf(float f) {
    union { float f; unsigned u; } x; x.f = f;
    unsigned u = x.u;
    unsigned r = (u + 0x7FFFu + ((u >> 16) & 1u)) >> 16;  // round-to-nearest-even
    return (unsigned short)r;
}

__device__ __forceinline__ void atomicMaxF(float* addr, float val) {
    int old = __float_as_int(*addr);
    while (val > __int_as_float(old)) {
        int prev = atomicCAS((int*)addr, old, __float_as_int(val));
        if (prev == old) break;
        old = prev;
    }
}

__device__ __forceinline__ void edge_sd(const int* ei, int e, int& s, int& d) {
    if (e < N_EDGES) { s = ei[e]; d = ei[N_EDGES + e]; }
    else             { s = d = e - N_EDGES; }            // appended self-loops
}

// ---------- conversion / transpose ----------
__global__ void k_f32_to_bf16(const float* __restrict__ in,
                              unsigned short* __restrict__ out, int n) {
    int i = blockIdx.x * blockDim.x + threadIdx.x;
    if (i < n) out[i] = f2bf(in[i]);
}

// W [K, Nout] row-major -> Wt [Nout, K] bf16
__global__ void k_transpose_bf16(const float* __restrict__ W,
                                 unsigned short* __restrict__ Wt, int K, int Nout) {
    int i = blockIdx.x * blockDim.x + threadIdx.x;
    if (i >= K * Nout) return;
    int k = i / Nout, n = i - k * Nout;
    Wt[n * K + k] = f2bf(W[i]);
}

__global__ void k_fill_f32(float* __restrict__ p, float v, int n) {
    int i = blockIdx.x * blockDim.x + threadIdx.x;
    if (i < n) p[i] = v;
}

// ---------- WMMA GEMM: C[M,Nout] = A[M,K](bf16) x Bt[Nout,K](bf16)^T ----------
// one wave -> 16 (M) x 64 (N) tile, k-step 32, f32 accumulate
__global__ __launch_bounds__(256)
void k_wmma_gemm(const unsigned short* __restrict__ A,
                 const unsigned short* __restrict__ Bt,
                 float* __restrict__ C, int M, int K, int Nout) {
    const int wave  = threadIdx.x >> 5;
    const int lane  = threadIdx.x & 31;
    const int mtile = blockIdx.x * 8 + wave;
    const int mTiles = (M + 15) >> 4;
    if (mtile >= mTiles) return;                 // wave-uniform: EXEC stays all-1s

    const int l16 = lane & 15;
    const int hf  = lane >> 4;                   // half-wave selects +8 K offset
    const int ntile0 = blockIdx.y * 64;

    const unsigned short* arow = A + (size_t)(mtile * 16 + l16) * K;
    const unsigned short* brow0 = Bt + (size_t)(ntile0 +  0 + l16) * K;
    const unsigned short* brow1 = Bt + (size_t)(ntile0 + 16 + l16) * K;
    const unsigned short* brow2 = Bt + (size_t)(ntile0 + 32 + l16) * K;
    const unsigned short* brow3 = Bt + (size_t)(ntile0 + 48 + l16) * K;

    v8f acc0 = {}, acc1 = {}, acc2 = {}, acc3 = {};
    union F { uint4 q[2]; v16bf v; };

    for (int kb = 0; kb < K; kb += 32) {
        const int eoff = kb + hf * 8;            // element offset (bf16)
        F af, b0, b1, b2, b3;
        { const uint4* p = (const uint4*)(arow  + eoff); af.q[0] = p[0]; af.q[1] = p[2]; }
        { const uint4* p = (const uint4*)(brow0 + eoff); b0.q[0] = p[0]; b0.q[1] = p[2]; }
        { const uint4* p = (const uint4*)(brow1 + eoff); b1.q[0] = p[0]; b1.q[1] = p[2]; }
        { const uint4* p = (const uint4*)(brow2 + eoff); b2.q[0] = p[0]; b2.q[1] = p[2]; }
        { const uint4* p = (const uint4*)(brow3 + eoff); b3.q[0] = p[0]; b3.q[1] = p[2]; }
        acc0 = __builtin_amdgcn_wmma_f32_16x16x32_bf16(false, af.v, false, b0.v, (short)0, acc0, false, false);
        acc1 = __builtin_amdgcn_wmma_f32_16x16x32_bf16(false, af.v, false, b1.v, (short)0, acc1, false, false);
        acc2 = __builtin_amdgcn_wmma_f32_16x16x32_bf16(false, af.v, false, b2.v, (short)0, acc2, false, false);
        acc3 = __builtin_amdgcn_wmma_f32_16x16x32_bf16(false, af.v, false, b3.v, (short)0, acc3, false, false);
    }

    // C/D layout: VGPR r -> row r (lanes 0-15) / 8+r (lanes 16-31); col = lane&15
    const int col0 = ntile0 + l16;
#pragma unroll
    for (int r = 0; r < 8; ++r) {
        int row = mtile * 16 + r + hf * 8;
        float* crow = C + (size_t)row * Nout + col0;
        crow[0]  = acc0[r];
        crow[16] = acc1[r];
        crow[32] = acc2[r];
        crow[48] = acc3[r];
    }
}

// ---------- attention scores: a_s[n,h] = <h[n,h,:], att_s[h,:]> ----------
__global__ void k_attn_scores(const float* __restrict__ h,
                              const float* __restrict__ atts,
                              const float* __restrict__ attd,
                              float* __restrict__ as_o, float* __restrict__ ad_o,
                              int H, int C) {
    int idx = blockIdx.x * blockDim.x + threadIdx.x;
    if (idx >= N_NODES * H) return;
    int n = idx / H, hd = idx - n * H;
    const float* hp = h + (size_t)n * H * C + hd * C;
    const float* sp = atts + hd * C;
    const float* dp = attd + hd * C;
    float s = 0.f, d = 0.f;
    for (int c = 0; c < C; ++c) { float v = hp[c]; s += v * sp[c]; d += v * dp[c]; }
    as_o[idx] = s; ad_o[idx] = d;
}

// ---------- edge pass A: segment max ----------
__global__ void k_edge_max(const int* __restrict__ ei,
                           const float* __restrict__ as_, const float* __restrict__ ad_,
                           float* __restrict__ emax, int H) {
    int idx = blockIdx.x * blockDim.x + threadIdx.x;
    if (idx >= E_TOT * H) return;
    int e = idx / H, hd = idx - e * H;
    int s, d; edge_sd(ei, e, s, d);
    float v = as_[s * H + hd] + ad_[d * H + hd];
    v = v > 0.f ? v : 0.2f * v;                  // leaky relu
    atomicMaxF(&emax[d * H + hd], v);
}

// ---------- edge pass B: exp + segment sum ----------
__global__ void k_edge_exp(const int* __restrict__ ei,
                           const float* __restrict__ as_, const float* __restrict__ ad_,
                           const float* __restrict__ emax,
                           float* __restrict__ exbuf, float* __restrict__ denom, int H) {
    int idx = blockIdx.x * blockDim.x + threadIdx.x;
    if (idx >= E_TOT * H) return;
    int e = idx / H, hd = idx - e * H;
    int s, d; edge_sd(ei, e, s, d);
    float v = as_[s * H + hd] + ad_[d * H + hd];
    v = v > 0.f ? v : 0.2f * v;
    float ex = __expf(v - emax[d * H + hd]);
    exbuf[idx] = ex;
    atomicAdd(&denom[d * H + hd], ex);
}

// ---------- edge pass C: weighted scatter (one wave per edge) ----------
__global__ __launch_bounds__(256)
void k_edge_agg(const int* __restrict__ ei, const float* __restrict__ h,
                const float* __restrict__ exbuf, const float* __restrict__ denom,
                float* __restrict__ agg, int H, int C) {
    int wid  = (blockIdx.x * blockDim.x + threadIdx.x) >> 5;
    int lane = threadIdx.x & 31;
    if (wid >= E_TOT) return;
    int s, d; edge_sd(ei, wid, s, d);
    const int HC = H * C;
    const float* hs = h + (size_t)s * HC;
    float* ag = agg + (size_t)d * HC;
    for (int ch = lane; ch < HC; ch += 32) {
        int hd = ch / C;
        float alpha = exbuf[wid * H + hd] / denom[d * H + hd];
        atomicAdd(&ag[ch], hs[ch] * alpha);
    }
}

// ---------- batchnorm stats (one block per channel) ----------
__global__ __launch_bounds__(256)
void k_bn_stats(const float* __restrict__ x, float* __restrict__ mean,
                float* __restrict__ rstd, int Cch) {
    int ch = blockIdx.x;
    float s = 0.f, s2 = 0.f;
    for (int n = threadIdx.x; n < N_NODES; n += blockDim.x) {
        float v = x[(size_t)n * Cch + ch];
        s += v; s2 += v * v;
    }
    __shared__ float sh[256], sh2[256];
    sh[threadIdx.x] = s; sh2[threadIdx.x] = s2;
    __syncthreads();
    for (int o = 128; o > 0; o >>= 1) {
        if (threadIdx.x < o) { sh[threadIdx.x] += sh[threadIdx.x + o];
                               sh2[threadIdx.x] += sh2[threadIdx.x + o]; }
        __syncthreads();
    }
    if (threadIdx.x == 0) {
        float mu  = sh[0] / (float)N_NODES;
        float var = sh2[0] / (float)N_NODES - mu * mu;
        mean[ch] = mu;
        rstd[ch] = rsqrtf(var + BN_EPS);
    }
}

// ---------- BN apply + ELU -> bf16 activations for next GEMM ----------
__global__ void k_bn_elu_bf16(const float* __restrict__ x,
                              const float* __restrict__ mean, const float* __restrict__ rstd,
                              const float* __restrict__ g, const float* __restrict__ b,
                              unsigned short* __restrict__ out, int Cch) {
    int i = blockIdx.x * blockDim.x + threadIdx.x;
    if (i >= N_NODES * Cch) return;
    int ch = i % Cch;
    float y = (x[i] - mean[ch]) * rstd[ch] * g[ch] + b[ch];
    y = y > 0.f ? y : (__expf(y) - 1.f);         // ELU
    out[i] = f2bf(y);
}

// ---------- classifier: relu(v@Wc1+bc1)@Wc2+bc2, v = agg2 + b2 ----------
__global__ void k_classifier(const float* __restrict__ agg2, const float* __restrict__ b2,
                             const float* __restrict__ Wc1, const float* __restrict__ bc1,
                             const float* __restrict__ Wc2, const float* __restrict__ bc2,
                             float* __restrict__ out) {
    int n = blockIdx.x * blockDim.x + threadIdx.x;
    if (n >= N_NODES) return;
    float v[64];
#pragma unroll
    for (int c = 0; c < 64; ++c) v[c] = agg2[(size_t)n * 64 + c] + b2[c];
    float o0 = bc2[0], o1 = bc2[1];
    for (int j = 0; j < 32; ++j) {
        float hsum = bc1[j];
#pragma unroll
        for (int c = 0; c < 64; ++c) hsum += v[c] * Wc1[c * 32 + j];
        hsum = hsum > 0.f ? hsum : 0.f;
        o0 += hsum * Wc2[j * 2 + 0];
        o1 += hsum * Wc2[j * 2 + 1];
    }
    out[n * 2 + 0] = o0;
    out[n * 2 + 1] = o1;
}

// ---------- host side ----------
static inline int cdiv(int a, int b) { return (a + b - 1) / b; }

extern "C" void kernel_launch(void* const* d_in, const int* in_sizes, int n_in,
                              void* d_out, int out_size, void* d_ws, size_t ws_size,
                              hipStream_t stream) {
    const float* x   = (const float*)d_in[0];
    const int*   ei  = (const int*)  d_in[1];
    const float* W0  = (const float*)d_in[2];
    const float* as0 = (const float*)d_in[3];
    const float* ad0 = (const float*)d_in[4];
    const float* W1  = (const float*)d_in[6];
    const float* as1 = (const float*)d_in[7];
    const float* ad1 = (const float*)d_in[8];
    const float* W2  = (const float*)d_in[10];
    const float* as2 = (const float*)d_in[11];
    const float* ad2 = (const float*)d_in[12];
    const float* b2  = (const float*)d_in[13];
    const float* g0  = (const float*)d_in[14];
    const float* be0 = (const float*)d_in[15];
    const float* g1  = (const float*)d_in[16];
    const float* be1 = (const float*)d_in[17];
    const float* Wc1 = (const float*)d_in[18];
    const float* bc1 = (const float*)d_in[19];
    const float* Wc2 = (const float*)d_in[20];
    const float* bc2 = (const float*)d_in[21];
    float* out = (float*)d_out;

    // workspace carve-out
    size_t off = 0;
    auto carve = [&](size_t bytes) -> void* {
        void* p = (char*)d_ws + off;
        off += (bytes + 255) & ~(size_t)255;
        return p;
    };
    float*          h      = (float*)         carve((size_t)N_NODES * HW * 4); // gemm out
    float*          agg    = (float*)         carve((size_t)N_NODES * HW * 4); // aggregated
    unsigned short* actbf  = (unsigned short*)carve((size_t)N_NODES * HW * 2); // bf16 gemm in
    unsigned short* wtbf   = (unsigned short*)carve((size_t)HW * HW * 2);      // bf16 W^T
    float*          as_b   = (float*)         carve((size_t)N_NODES * HEADS * 4);
    float*          ad_b   = (float*)         carve((size_t)N_NODES * HEADS * 4);
    float*          emax   = (float*)         carve((size_t)N_NODES * HEADS * 4);
    float*          denom  = (float*)         carve((size_t)N_NODES * HEADS * 4);
    float*          exbuf  = (float*)         carve((size_t)E_TOT * HEADS * 4);
    float*          mean   = (float*)         carve(HW * 4);
    float*          rstd   = (float*)         carve(HW * 4);

    const int B = 256;
    const int mTiles  = N_NODES / 16;          // 3125
    const int mBlocks = cdiv(mTiles, 8);       // 8 waves per block

    struct Layer {
        const float *W, *atts, *attd, *g, *be;
        int K, Nout, H, C;
        bool bn;
    };
    Layer layers[3] = {
        { W0, as0, ad0, g0, be0, IN_CH, HW,  HEADS, HID, true  },
        { W1, as1, ad1, g1, be1, HW,    HW,  HEADS, HID, true  },
        { W2, as2, ad2, 0,  0,   HW,    HID, 1,     HID, false },
    };

    // layer-0 GEMM input: x -> bf16
    k_f32_to_bf16<<<cdiv(N_NODES * IN_CH, B), B, 0, stream>>>(x, actbf, N_NODES * IN_CH);

    for (int li = 0; li < 3; ++li) {
        const Layer& L = layers[li];
        const int HC = L.H * L.C;

        // weights: transpose+convert
        k_transpose_bf16<<<cdiv(L.K * L.Nout, B), B, 0, stream>>>(L.W, wtbf, L.K, L.Nout);

        // h = act @ W  (WMMA)
        dim3 ggrid(mBlocks, L.Nout / 64);
        k_wmma_gemm<<<ggrid, B, 0, stream>>>(actbf, wtbf, h, N_NODES, L.K, L.Nout);

        // attention scores
        k_attn_scores<<<cdiv(N_NODES * L.H, B), B, 0, stream>>>(h, L.atts, L.attd,
                                                                as_b, ad_b, L.H, L.C);

        // init accumulators
        k_fill_f32<<<cdiv(N_NODES * L.H, B), B, 0, stream>>>(emax, -1e30f, N_NODES * L.H);
        hipMemsetAsync(denom, 0, (size_t)N_NODES * L.H * 4, stream);
        hipMemsetAsync(agg,   0, (size_t)N_NODES * HC * 4, stream);

        // softmax + scatter
        k_edge_max<<<cdiv(E_TOT * L.H, B), B, 0, stream>>>(ei, as_b, ad_b, emax, L.H);
        k_edge_exp<<<cdiv(E_TOT * L.H, B), B, 0, stream>>>(ei, as_b, ad_b, emax,
                                                           exbuf, denom, L.H);
        k_edge_agg<<<cdiv(E_TOT, 8), B, 0, stream>>>(ei, h, exbuf, denom, agg, L.H, L.C);

        if (L.bn) {
            // bias is zero AND cancels inside BN; BN + ELU -> bf16 for next GEMM
            k_bn_stats<<<HC, B, 0, stream>>>(agg, mean, rstd, HC);
            k_bn_elu_bf16<<<cdiv(N_NODES * HC, B), B, 0, stream>>>(agg, mean, rstd,
                                                                   L.g, L.be, actbf, HC);
        }
    }

    // classifier (b2 folded in here)
    k_classifier<<<cdiv(N_NODES, B), B, 0, stream>>>(agg, b2, Wc1, bc1, Wc2, bc2, out);
}
